// STSA_59253368816112
// MI455X (gfx1250) — matrix operand order repaired and checked by reading, not compiled, not run on hardware
//
#include <hip/hip_runtime.h>
#include <stdint.h>

// Problem constants
#define BB 4
#define TT 24
#define MM 512
#define DD 128
#define HH 8
#define KN 16            // neighbors per node
#define DH 16
#define NROWS (BB*TT*MM) // 49152
#define SCL 0.25f        // 1/sqrt(DH)

// WMMA vector types (wave32, gfx1250)
typedef __attribute__((ext_vector_type(16))) __bf16 v16bf;
typedef __attribute__((ext_vector_type(8)))  float  v8f;

union FragAB { v16bf v; unsigned short u[16]; };

// Native bf16 converts (RNE) — let clang pick v_cvt ops instead of manual shifts
__device__ __forceinline__ unsigned short f2bf(float f) {
  union { __bf16 b; unsigned short u; } x; x.b = (__bf16)f; return x.u;
}
__device__ __forceinline__ float bf2f(unsigned short h) {
  union { unsigned short u; __bf16 b; } x; x.u = h; return (float)x.b;
}

// ---------------------------------------------------------------------------
// Async global->LDS path (CDNA5): guarded so either toolchain still compiles.
// Builtin signature (from diagnostics): (v4i addrspace(1)*, v4i addrspace(3)*,
// imm offset, imm cpol). AS3 pointers are 32-bit; low 32 bits of a flat LDS
// address are the LDS byte address (ISA aperture rule), so integer casts give
// valid AS pointers.
// ---------------------------------------------------------------------------
#if __has_builtin(__builtin_amdgcn_global_load_async_to_lds_b128) && \
    __has_builtin(__builtin_amdgcn_s_wait_asynccnt)
#define USE_ASYNC_LDS 1
typedef __attribute__((ext_vector_type(4))) int v4i_t;
typedef __attribute__((address_space(1))) v4i_t gv4i;
typedef __attribute__((address_space(3))) v4i_t lv4i;
__device__ __forceinline__ void async_cp16B(const unsigned short* g, unsigned short* l) {
  __builtin_amdgcn_global_load_async_to_lds_b128(
      (gv4i*)(uintptr_t)g, (lv4i*)(unsigned int)(uintptr_t)l, 0, 0);
}
__device__ __forceinline__ void async_wait0() { __builtin_amdgcn_s_wait_asynccnt(0); }
#else
#define USE_ASYNC_LDS 0
__device__ __forceinline__ void async_cp16B(const unsigned short* g, unsigned short* l) {
#pragma unroll
  for (int e = 0; e < 8; ++e) l[e] = g[e];
}
__device__ __forceinline__ void async_wait0() {}
#endif

// ---------------------------------------------------------------------------
// K0: pack concatenated bf16 weight matrices + fused bias vector
//   Wcat  [128][768] = [t_Wqkv(384) | s_Wq(128) | s_Wkv(256)]
//   Wocat [256][128] = [t_Wo ; s_Wo]
//   biasCat[768]     = [t_bqkv | s_bq | s_bkv]
// ---------------------------------------------------------------------------
__global__ void stsa_prep(const float* __restrict__ tW, const float* __restrict__ sW,
                          const float* __restrict__ tB, const float* __restrict__ sB,
                          const float* __restrict__ tWo, const float* __restrict__ sWo,
                          unsigned short* __restrict__ Wcat,
                          unsigned short* __restrict__ Wocat,
                          float* __restrict__ biasCat) {
  int i = blockIdx.x * blockDim.x + threadIdx.x;
  if (i < 128 * 768) {
    int k = i / 768, n = i % 768;
    float w;
    if (n < 384)      w = tW[k * 384 + n];
    else if (n < 512) w = sW[k * 384 + (n - 384)];
    else              w = sW[k * 384 + 128 + (n - 512)];
    Wcat[i] = f2bf(w);
  }
  int j = i - 128 * 768;
  if (j >= 0 && j < 256 * 128) {
    int k = j / 128, n = j % 128;
    float w = (k < 128) ? tWo[k * 128 + n] : sWo[(k - 128) * 128 + n];
    Wocat[j] = f2bf(w);
  }
  int l = i - 128 * 768 - 256 * 128;
  if (l >= 0 && l < 768) {
    float b;
    if (l < 384)      b = tB[l];
    else if (l < 512) b = sB[l - 384];
    else              b = sB[128 + (l - 512)];
    biasCat[l] = b;
  }
}

// ---------------------------------------------------------------------------
// Per-lane WMMA fragment packers (wave32 ISA layouts, 16-bit data)
//   A 16x32: lane m=lane&15; koff=(lane>>4)*8; elems 0..7 -> K=koff+e,
//            elems 8..15 -> K=16+koff+e
//   B 32x16: lane n=lane&15; elems 0..15 -> K in the same order as A
// ---------------------------------------------------------------------------
__device__ __forceinline__ void packA(FragAB& f, const unsigned short* row,
                                      int koff) {
#pragma unroll
  for (int e = 0; e < 8; ++e) {
    f.u[e]     = row[koff + e];
    f.u[8 + e] = row[16 + koff + e];
  }
}

// ---------------------------------------------------------------------------
// K1: GEMM  P[49152 x 768] = bf16(X[49152 x 128]) * Wcat[128 x 768] + biasCat
//     block 256 thr, tile 128x64, 8 waves in 4x2 -> each wave 32x32 (2x2 WMMA)
//     A tile: sync load fp32 + v_cvt to bf16; B tile: async global->LDS
// ---------------------------------------------------------------------------
__global__ __launch_bounds__(256)
void stsa_gemm1(const float* __restrict__ X, const unsigned short* __restrict__ Wcat,
                const float* __restrict__ biasCat, unsigned short* __restrict__ P) {
  __shared__ unsigned short As[128][40];  // row pitch 80B (16B-aligned)
  __shared__ unsigned short Bs[32][72];   // row pitch 144B (16B-aligned)
  const int tid  = threadIdx.x;
  const int lane = tid & 31;
  const int wave = tid >> 5;
  const int rowBase = blockIdx.x * 128;
  const int colBase = blockIdx.y * 64;
  const int wr = (wave >> 1) * 32;
  const int wc = (wave & 1) * 32;

  v8f acc[2][2];
#pragma unroll
  for (int a = 0; a < 2; ++a)
#pragma unroll
    for (int b = 0; b < 2; ++b) acc[a][b] = (v8f){0,0,0,0,0,0,0,0};

  for (int k0 = 0; k0 < 128; k0 += 32) {
    // B tile: 32x64 bf16, one 16B async chunk per thread
    {
      const int k  = tid >> 3;
      const int c0 = (tid & 7) * 8;
      async_cp16B(Wcat + (size_t)(k0 + k) * 768 + colBase + c0, &Bs[k][c0]);
    }
    // A tile: 128x32 fp32 -> bf16 (each thread: 16 contiguous floats)
    {
      const int i  = tid >> 1;
      const int j0 = (tid & 1) * 16;
      const float* src = X + (size_t)(rowBase + i) * 128 + k0 + j0;
      if (k0 + 32 < 128) __builtin_prefetch(src + 32, 0, 0); // global_prefetch_b8
#pragma unroll
      for (int e = 0; e < 16; ++e) As[i][j0 + e] = f2bf(src[e]);
    }
    async_wait0();
    __syncthreads();

    FragAB af[2], bf[2];
    const int m    = lane & 15;
    const int koff = (lane >> 4) * 8;
#pragma unroll
    for (int t = 0; t < 2; ++t) packA(af[t], &As[wr + t * 16 + m][0], koff);
#pragma unroll
    for (int t = 0; t < 2; ++t) {
#pragma unroll
      for (int e = 0; e < 16; ++e) bf[t].u[e] = Bs[koff + (e & 7) + ((e >> 3) << 4)][wc + t * 16 + m];
    }
#pragma unroll
    for (int a = 0; a < 2; ++a)
#pragma unroll
      for (int b = 0; b < 2; ++b)
        acc[a][b] = __builtin_amdgcn_wmma_f32_16x16x32_bf16(
            false, af[a].v, false, bf[b].v, (short)0, acc[a][b], false, false);
    __syncthreads();
  }

  // epilogue: C/D layout -> VGPR i: row = i + 8*(lane>=16), col = lane&15
  const int rOff = (lane >> 4) * 8;
  const int n    = lane & 15;
#pragma unroll
  for (int a = 0; a < 2; ++a)
#pragma unroll
    for (int b = 0; b < 2; ++b)
#pragma unroll
      for (int i = 0; i < 8; ++i) {
        const int g   = rowBase + wr + a * 16 + rOff + i;
        const int col = colBase + wc + b * 16 + n;
        P[(size_t)g * 768 + col] = f2bf(acc[a][b][i] + biasCat[col]);
      }
}

// ---------------------------------------------------------------------------
// K2: temporal attention, one wave per (b, m, head); T=24, DH=16
//     P cols: t_q = [0,128), t_k = [128,256), t_v = [256,384)  (head h -> h*16)
//     writes AO cols [0,128)
// ---------------------------------------------------------------------------
__global__ __launch_bounds__(256)
void stsa_tattn(const unsigned short* __restrict__ P, unsigned short* __restrict__ AO) {
  __shared__ unsigned short qs[HH][TT * DH];
  __shared__ unsigned short ks[HH][TT * DH];
  __shared__ unsigned short vs[HH][TT * DH];
  __shared__ float          Ss[HH][TT * TT];
  const int bm   = blockIdx.x;  // b*512 + m
  const int b    = bm >> 9;
  const int m    = bm & 511;
  const int h    = threadIdx.x >> 5;
  const int lane = threadIdx.x & 31;

  for (int idx = lane; idx < TT * DH; idx += 32) {
    const int t = idx >> 4, e = idx & 15;
    const size_t base = (size_t)((b * TT + t) * MM + m) * 768 + h * 16 + e;
    qs[h][idx] = P[base];
    ks[h][idx] = P[base + 128];
    vs[h][idx] = P[base + 256];
  }
  __syncthreads();
  for (int p = lane; p < TT * TT; p += 32) {
    const int r = p / TT, c = p % TT;
    float s = 0.f;
#pragma unroll
    for (int e = 0; e < DH; ++e)
      s += bf2f(qs[h][r * 16 + e]) * bf2f(ks[h][c * 16 + e]);
    Ss[h][p] = s * SCL;
  }
  __syncthreads();
  if (lane < TT) {
    const int r = lane;
    float mx = -1e30f;
    for (int c = 0; c < TT; ++c) mx = fmaxf(mx, Ss[h][r * TT + c]);
    float sum = 0.f;
    for (int c = 0; c < TT; ++c) {
      const float ev = __expf(Ss[h][r * TT + c] - mx);
      Ss[h][r * TT + c] = ev; sum += ev;
    }
    const float inv = 1.0f / sum;
    for (int c = 0; c < TT; ++c) Ss[h][r * TT + c] *= inv;
  }
  __syncthreads();
  for (int p = lane; p < TT * DH; p += 32) {
    const int t = p >> 4, e = p & 15;
    float o = 0.f;
    for (int c = 0; c < TT; ++c) o += Ss[h][t * TT + c] * bf2f(vs[h][c * 16 + e]);
    AO[(size_t)((b * TT + t) * MM + m) * 256 + h * 16 + e] = f2bf(o);
  }
}

// ---------------------------------------------------------------------------
// K3: spatial attention; gather-after-project (shared linear map => legal).
//     one block of 128 thr per row r=(b,t,m); thread = (h, x) with x = kk or e.
//     P cols: s_q = [384,512), s_k = [512,640), s_v = [640,768)
//     writes AO cols [128,256) (zeroed when deg==0)
// ---------------------------------------------------------------------------
__global__ __launch_bounds__(128)
void stsa_sattn(const unsigned short* __restrict__ P, const int* __restrict__ neighbors,
                const int* __restrict__ deg, unsigned short* __restrict__ AO) {
  __shared__ unsigned short qL[128];
  __shared__ unsigned short kL[KN][128];
  __shared__ unsigned short vL[KN][128];
  __shared__ float          sL[HH][KN];
  const int r      = blockIdx.x;      // (b*T+t)*M + m
  const int m      = r % MM;
  const int btBase = (r / MM) * MM;
  const int tid    = threadIdx.x;
  const int h      = tid >> 4;
  const int x      = tid & 15;
  const int dg     = deg[m];

  qL[tid] = P[(size_t)r * 768 + 384 + tid];
  {
    const int kk = x;
    const int nb = neighbors[m * KN + kk];
    const size_t base = (size_t)(btBase + nb) * 768 + 512 + h * 16;
#pragma unroll
    for (int e = 0; e < 16; ++e) {
      kL[kk][h * 16 + e] = P[base + e];
      vL[kk][h * 16 + e] = P[base + 128 + e];
    }
  }
  __syncthreads();
  {
    const int kk = x;
    float s = 0.f;
#pragma unroll
    for (int e = 0; e < 16; ++e) s += bf2f(qL[h * 16 + e]) * bf2f(kL[kk][h * 16 + e]);
    sL[h][kk] = (kk < dg) ? s * SCL : -1e9f;
  }
  __syncthreads();
  float mx = -1e30f;
  for (int j = 0; j < KN; ++j) mx = fmaxf(mx, sL[h][j]);
  float sum = 0.f;
  for (int j = 0; j < KN; ++j) sum += __expf(sL[h][j] - mx);
  const float a = __expf(sL[h][x] - mx) / sum;
  __syncthreads();
  sL[h][x] = a;
  __syncthreads();
  {
    const int e = x;
    float o = 0.f;
    for (int kk = 0; kk < KN; ++kk) o += sL[h][kk] * bf2f(vL[kk][h * 16 + e]);
    if (dg == 0) o = 0.f;
    AO[(size_t)r * 256 + 128 + h * 16 + e] = f2bf(o);
  }
}

// ---------------------------------------------------------------------------
// K4: GEMM  Out[49152 x 128] = AO[49152 x 256] * Wocat[256 x 128]
//            + t_bo + (deg[m]>0 ? s_bo : 0)
//     Both tiles already bf16 in memory -> fully async global->LDS staging
// ---------------------------------------------------------------------------
__global__ __launch_bounds__(256)
void stsa_gemm2(const unsigned short* __restrict__ AO, const unsigned short* __restrict__ Wocat,
                const float* __restrict__ t_bo, const float* __restrict__ s_bo,
                const int* __restrict__ deg, float* __restrict__ Out) {
  __shared__ unsigned short As[128][40];
  __shared__ unsigned short Bs[32][72];
  const int tid  = threadIdx.x;
  const int lane = tid & 31;
  const int wave = tid >> 5;
  const int rowBase = blockIdx.x * 128;
  const int colBase = blockIdx.y * 64;
  const int wr = (wave >> 1) * 32;
  const int wc = (wave & 1) * 32;

  v8f acc[2][2];
#pragma unroll
  for (int a = 0; a < 2; ++a)
#pragma unroll
    for (int b = 0; b < 2; ++b) acc[a][b] = (v8f){0,0,0,0,0,0,0,0};

  for (int k0 = 0; k0 < 256; k0 += 32) {
    // A tile: 128x32 bf16, two 16B async chunks per thread
    {
      const int i  = tid >> 1;
      const int j0 = (tid & 1) * 16;
      const unsigned short* src = AO + (size_t)(rowBase + i) * 256 + k0 + j0;
      if (k0 + 32 < 256) __builtin_prefetch(src + 32, 0, 0);
      async_cp16B(src,     &As[i][j0]);
      async_cp16B(src + 8, &As[i][j0 + 8]);
    }
    // B tile: 32x64 bf16, one 16B async chunk per thread
    {
      const int k  = tid >> 3;
      const int c0 = (tid & 7) * 8;
      async_cp16B(Wocat + (size_t)(k0 + k) * 128 + colBase + c0, &Bs[k][c0]);
    }
    async_wait0();
    __syncthreads();

    FragAB af[2], bf[2];
    const int m    = lane & 15;
    const int koff = (lane >> 4) * 8;
#pragma unroll
    for (int t = 0; t < 2; ++t) packA(af[t], &As[wr + t * 16 + m][0], koff);
#pragma unroll
    for (int t = 0; t < 2; ++t) {
#pragma unroll
      for (int e = 0; e < 16; ++e) bf[t].u[e] = Bs[koff + (e & 7) + ((e >> 3) << 4)][wc + t * 16 + m];
    }
#pragma unroll
    for (int a = 0; a < 2; ++a)
#pragma unroll
      for (int b = 0; b < 2; ++b)
        acc[a][b] = __builtin_amdgcn_wmma_f32_16x16x32_bf16(
            false, af[a].v, false, bf[b].v, (short)0, acc[a][b], false, false);
    __syncthreads();
  }

  const int rOff = (lane >> 4) * 8;
  const int n    = lane & 15;
#pragma unroll
  for (int a = 0; a < 2; ++a)
#pragma unroll
    for (int b = 0; b < 2; ++b)
#pragma unroll
      for (int i = 0; i < 8; ++i) {
        const int g   = rowBase + wr + a * 16 + rOff + i;
        const int col = colBase + wc + b * 16 + n;
        const int mm  = g % MM;
        const float bias = t_bo[col] + ((deg[mm] > 0) ? s_bo[col] : 0.f);
        Out[(size_t)g * 128 + col] = acc[a][b][i] + bias;
      }
}

// ---------------------------------------------------------------------------
extern "C" void kernel_launch(void* const* d_in, const int* in_sizes, int n_in,
                              void* d_out, int out_size, void* d_ws, size_t ws_size,
                              hipStream_t stream) {
  (void)in_sizes; (void)n_in; (void)out_size; (void)ws_size;
  const float* x       = (const float*)d_in[0];
  const float* t_Wqkv  = (const float*)d_in[1];
  const float* t_bqkv  = (const float*)d_in[2];
  const float* t_Wo    = (const float*)d_in[3];
  const float* t_bo    = (const float*)d_in[4];
  const float* s_Wqkv  = (const float*)d_in[5];
  const float* s_bqkv  = (const float*)d_in[6];
  const float* s_Wo    = (const float*)d_in[7];
  const float* s_bo    = (const float*)d_in[8];
  const int*   neigh   = (const int*)d_in[9];
  const int*   deg     = (const int*)d_in[10];

  // workspace layout (bytes)
  uint8_t* ws = (uint8_t*)d_ws;
  const size_t offP    = 0;                                   // 49152*768*2  = 75,497,472
  const size_t offAO   = offP    + (size_t)NROWS * 768 * 2;   // 49152*256*2  = 25,165,824
  const size_t offWcat = offAO   + (size_t)NROWS * 256 * 2;   // 128*768*2
  const size_t offWoc  = offWcat + (size_t)128 * 768 * 2;     // 256*128*2
  const size_t offBias = offWoc  + (size_t)256 * 128 * 2;     // 768*4
  unsigned short* P     = (unsigned short*)(ws + offP);
  unsigned short* AO    = (unsigned short*)(ws + offAO);
  unsigned short* Wcat  = (unsigned short*)(ws + offWcat);
  unsigned short* Wocat = (unsigned short*)(ws + offWoc);
  float*          biasC = (float*)(ws + offBias);

  stsa_prep<<<516, 256, 0, stream>>>(t_Wqkv, s_Wqkv, t_bqkv, s_bqkv, t_Wo, s_Wo,
                                     Wcat, Wocat, biasC);
  stsa_gemm1<<<dim3(NROWS / 128, 12), 256, 0, stream>>>(x, Wcat, biasC, P);
  stsa_tattn<<<BB * MM, 256, 0, stream>>>(P, AO);
  stsa_sattn<<<NROWS, 128, 0, stream>>>(P, neigh, deg, AO);
  stsa_gemm2<<<dim3(NROWS / 128, 2), 256, 0, stream>>>(AO, Wocat, t_bo, s_bo, deg,
                                                       (float*)d_out);
}
